// SparseConvCausalAttention_32255204393234
// MI455X (gfx1250) — matrix operand order repaired and checked by reading, not compiled, not run on hardware
//
#include <hip/hip_runtime.h>

typedef __attribute__((ext_vector_type(16))) _Float16 v16h;
typedef __attribute__((ext_vector_type(8)))  float    v8f;

#define B_     4
#define NREAL  1151
#define NP     1152
#define DIM_   512
#define H_     8
#define DH_    64
#define TEXT   128
#define IMGSEQ 1024
#define IMGW   32
#define NEG_BIG (-3.0e38f)

__device__ __forceinline__ v8f wmma16(v16h a, v16h b, v8f c) {
  return __builtin_amdgcn_wmma_f32_16x16x32_f16(
      /*neg_a=*/false, a, /*neg_b=*/false, b,
      /*c_mod=*/(short)0, c, /*reuse_a=*/false, /*reuse_b=*/false);
}

// Table-order A-fragment load (p[0..7], p[16..23]) (row-major M x K, 16x32 tile)
__device__ __forceinline__ v16h load_a(const _Float16* __restrict__ p) {
  v16h a;
#pragma unroll
  for (int e = 0; e < 8; ++e) { a[e] = p[e]; a[e + 8] = p[16 + e]; }
  return a;
}
// Table-order B-fragment load (p[0..7], p[16..23]) from K-contiguous (N-major / transposed) storage
__device__ __forceinline__ v16h load_b(const _Float16* __restrict__ p) {
  v16h b;
#pragma unroll
  for (int e = 0; e < 8; ++e) { b[e] = p[e]; b[e + 8] = p[16 + e]; }
  return b;
}

// ---------------------------------------------------------------- converts
__global__ void cvt_x_pad(const float* __restrict__ x, _Float16* __restrict__ xh) {
  int idx = blockIdx.x * 256 + threadIdx.x;          // over B_*NP*DIM_
  if (idx >= B_ * NP * DIM_) return;
  int c = idx & (DIM_ - 1);
  int t = (idx >> 9) % NP;
  int b = idx / (NP * DIM_);
  float v = (t < NREAL) ? x[(b * NREAL + t) * DIM_ + c] : 0.0f;
  xh[idx] = (_Float16)v;
}

// w (K x N) f32 row-major  ->  wT (N x K) f16
__global__ void cvt_wT(const float* __restrict__ w, _Float16* __restrict__ wT, int K, int N) {
  int idx = blockIdx.x * 256 + threadIdx.x;          // over N*K, k fastest
  if (idx >= K * N) return;
  int k = idx % K, n = idx / K;
  wT[idx] = (_Float16)w[k * N + n];
}

// ---------------------------------------------------------------- QKV GEMM
// M=4608, N=1536, K=512. grid.y selects 128-col group -> q|k|v section is
// block-uniform (128 | 512); M-tile never straddles a batch (1152 % 16 == 0).
__global__ void qkv_gemm(const _Float16* __restrict__ xh, const _Float16* __restrict__ wqT,
                         _Float16* __restrict__ qh, _Float16* __restrict__ kh,
                         _Float16* __restrict__ vh, _Float16* __restrict__ vTh) {
  int wave = threadIdx.x >> 5, lane = threadIdx.x & 31;
  int hi = lane >> 4, ln = lane & 15;
  int m0 = blockIdx.x * 16;
  int n0 = blockIdx.y * 128 + wave * 16;
  v8f z = {}; v8f acc = z;
  const _Float16* arow = xh + (m0 + ln) * DIM_ + hi * 8;
  const _Float16* brow = wqT + (n0 + ln) * DIM_ + hi * 8;
  for (int kc = 0; kc < DIM_; kc += 32) {
    __builtin_prefetch(arow + kc + 128, 0, 3);
    __builtin_prefetch(brow + kc + 128, 0, 3);
    acc = wmma16(load_a(arow + kc), load_b(brow + kc), acc);
  }
  // uniform epilogue factors
  int b    = m0 / NP;                  // wave-uniform batch
  int tt0  = m0 - b * NP;              // row base within batch
  int sec  = n0 >> 9;                  // 0=q 1=k 2=v (block-uniform)
  int inner = (n0 & 511) + ln;
  int h = inner >> 6, d = inner & 63;
  int bh = b * H_ + h;
  if (sec == 0) {
#pragma unroll
    for (int e = 0; e < 8; ++e)
      qh[(bh * NP + tt0 + e + hi * 8) * DH_ + d] = (_Float16)(acc[e] * 0.125f);
  } else if (sec == 1) {
#pragma unroll
    for (int e = 0; e < 8; ++e)
      kh[(bh * NP + tt0 + e + hi * 8) * DH_ + d] = (_Float16)acc[e];
  } else {
#pragma unroll
    for (int e = 0; e < 8; ++e) {
      int tt = tt0 + e + hi * 8;
      _Float16 hv = (_Float16)acc[e];
      vh[(bh * NP + tt) * DH_ + d] = hv;
      vTh[(bh * DH_ + d) * NP + tt] = hv;
    }
  }
}

// ---------------------------------------------------------------- text attention
// One block per (b,h): 8 waves x 16 rows = 128 text rows. Causal softmax.
__global__ void text_attn(const _Float16* __restrict__ qh, const _Float16* __restrict__ kh,
                          const _Float16* __restrict__ vTh, _Float16* __restrict__ innerh) {
  __shared__ _Float16 probs[8][16 * TEXT];           // 32 KB
  int bh = blockIdx.x, b = bh >> 3, h = bh & 7;
  int wave = threadIdx.x >> 5, lane = threadIdx.x & 31;
  int hi = lane >> 4, ln = lane & 15;
  int r0 = wave * 16;
  const _Float16* Q  = qh  + bh * NP * DH_;
  const _Float16* K  = kh  + bh * NP * DH_;
  const _Float16* VT = vTh + bh * DH_ * NP;          // [d][t]

  v8f z = {}; v8f acc[8];
#pragma unroll
  for (int j = 0; j < 8; ++j) acc[j] = z;
  for (int kc = 0; kc < DH_; kc += 32) {
    v16h a = load_a(Q + (r0 + ln) * DH_ + hi * 8 + kc);
#pragma unroll
    for (int j = 0; j < 8; ++j)
      acc[j] = wmma16(a, load_b(K + (j * 16 + ln) * DH_ + kc + hi * 8), acc[j]);
  }
  // causal softmax per row (rows live in 16-lane halves of C layout)
#pragma unroll
  for (int e = 0; e < 8; ++e) {
    int r = r0 + e + hi * 8;                         // query row 0..127
    float vals[8], mx = NEG_BIG;
#pragma unroll
    for (int j = 0; j < 8; ++j) {
      int c = j * 16 + ln;
      float v = (c <= r) ? acc[j][e] : NEG_BIG;
      vals[j] = v; mx = fmaxf(mx, v);
    }
    for (int s = 1; s < 16; s <<= 1) mx = fmaxf(mx, __shfl_xor(mx, s));
    float sum = 0.0f;
#pragma unroll
    for (int j = 0; j < 8; ++j) { vals[j] = __expf(vals[j] - mx); sum += vals[j]; }
    for (int s = 1; s < 16; s <<= 1) sum += __shfl_xor(sum, s);
    float inv = 1.0f / sum;
#pragma unroll
    for (int j = 0; j < 8; ++j)
      probs[wave][(e + hi * 8) * TEXT + j * 16 + ln] = (_Float16)(vals[j] * inv);
  }
  __syncthreads();
  // attn @ V_text : M=16, K=128, N=64 ; B from VT (K-contiguous)
  v8f oacc[4];
#pragma unroll
  for (int j = 0; j < 4; ++j) oacc[j] = z;
  for (int kc = 0; kc < TEXT; kc += 32) {
    v16h a = load_a(&probs[wave][ln * TEXT + kc + hi * 8]);
#pragma unroll
    for (int j = 0; j < 4; ++j)
      oacc[j] = wmma16(a, load_b(VT + (j * 16 + ln) * NP + kc + hi * 8), oacc[j]);
  }
#pragma unroll
  for (int j = 0; j < 4; ++j)
#pragma unroll
    for (int e = 0; e < 8; ++e) {
      int t = r0 + e + hi * 8;                       // text row == Np row
      innerh[(b * NP + t) * DIM_ + h * DH_ + j * 16 + ln] = (_Float16)oacc[j][e];
    }
}

// ---------------------------------------------------------------- image attention
// grid (32 bh, 16 rowgroups of 64 image rows), 128 threads (4 waves x 16 rows).
__global__ void img_attn(const _Float16* __restrict__ qh, const _Float16* __restrict__ kh,
                         const _Float16* __restrict__ vh, const _Float16* __restrict__ vTh,
                         const unsigned char* __restrict__ mask,
                         _Float16* __restrict__ innerh) {
  __shared__ float lg[64][160];                      // 40 KB: [row][0..127 text | 128..152 window]
  int bh = blockIdx.x, b = bh >> 3, h = bh & 7;
  int wave = threadIdx.x >> 5, lane = threadIdx.x & 31;
  int hi = lane >> 4, ln = lane & 15;
  int r0 = wave * 16;
  int i0 = blockIdx.y * 64;                          // first image row of block
  const _Float16* Qi = qh + bh * NP * DH_ + TEXT * DH_;
  const _Float16* Kt = kh + bh * NP * DH_;
  const _Float16* Ki = Kt + TEXT * DH_;
  const _Float16* VT = vTh + bh * DH_ * NP;          // [d][t], text cols 0..127
  const _Float16* Vi = vh + bh * NP * DH_ + TEXT * DH_;

  // Phase A: dots_it = Q_img strip @ K_text^T  (WMMA)
  v8f z = {}; v8f acc[8];
#pragma unroll
  for (int j = 0; j < 8; ++j) acc[j] = z;
  for (int kc = 0; kc < DH_; kc += 32) {
    v16h a = load_a(Qi + (i0 + r0 + ln) * DH_ + hi * 8 + kc);
#pragma unroll
    for (int j = 0; j < 8; ++j)
      acc[j] = wmma16(a, load_b(Kt + (j * 16 + ln) * DH_ + kc + hi * 8), acc[j]);
  }
#pragma unroll
  for (int j = 0; j < 8; ++j)
#pragma unroll
    for (int e = 0; e < 8; ++e) {
      int lr = r0 + e + hi * 8, col = j * 16 + ln;
      lg[lr][col] = mask[b * TEXT + col] ? acc[j][e] : NEG_BIG;
    }
  // Phase B: 5x5 window dots, vectorized b128 q/k row loads
  for (int item = threadIdx.x; item < 64 * 25; item += 128) {
    int lr = item / 25, w = item % 25;
    int i = i0 + lr, iy = i >> 5, ix = i & 31;
    int ky = iy + w / 5 - 2, kx = ix + (w % 5) - 2;
    int jpos = ky * IMGW + kx;
    float lv = NEG_BIG;
    if (ky >= 0 && ky < IMGW && kx >= 0 && kx < IMGW && jpos <= i) {
      const v16h* qp = (const v16h*)(Qi + i * DH_);     // 128B-aligned rows
      const v16h* kp = (const v16h*)(Ki + jpos * DH_);
      float s = 0.0f;
#pragma unroll
      for (int c = 0; c < DH_ / 16; ++c) {
        v16h qa = qp[c], ka = kp[c];
#pragma unroll
        for (int e = 0; e < 16; ++e) s += (float)qa[e] * (float)ka[e];
      }
      lv = s;
    }
    lg[lr][TEXT + w] = lv;
  }
  __syncthreads();
  // Phase C: fused softmax over 153 logits, normalize in place
  if (threadIdx.x < 64) {
    int lr = threadIdx.x;
    float mx = NEG_BIG;
    for (int c = 0; c < 153; ++c) mx = fmaxf(mx, lg[lr][c]);
    float sum = 0.0f;
    for (int c = 0; c < 153; ++c) { float e = __expf(lg[lr][c] - mx); lg[lr][c] = e; sum += e; }
    float inv = 1.0f / sum;
    for (int c = 0; c < 153; ++c) lg[lr][c] *= inv;
  }
  __syncthreads();
  // Phase D: out = a_t @ V_text (WMMA) + window accumulation (VALU)
  v8f oacc[4];
#pragma unroll
  for (int j = 0; j < 4; ++j) oacc[j] = z;
  for (int kc = 0; kc < TEXT; kc += 32) {
    v16h a;
#pragma unroll
    for (int e = 0; e < 8; ++e) {
      a[e]     = (_Float16)lg[r0 + ln][kc + hi * 8 + e];
      a[e + 8] = (_Float16)lg[r0 + ln][kc + hi * 8 + 16 + e];
    }
#pragma unroll
    for (int j = 0; j < 4; ++j)
      oacc[j] = wmma16(a, load_b(VT + (j * 16 + ln) * NP + kc + hi * 8), oacc[j]);
  }
#pragma unroll
  for (int e = 0; e < 8; ++e) {
    int lr = r0 + e + hi * 8;
    int i = i0 + lr, iy = i >> 5, ix = i & 31;
    for (int w = 0; w < 25; ++w) {
      int ky = iy + w / 5 - 2, kx = ix + (w % 5) - 2;
      if (ky < 0 || ky >= IMGW || kx < 0 || kx >= IMGW) continue;
      int jpos = ky * IMGW + kx;
      if (jpos > i) continue;
      float p = lg[lr][TEXT + w];
      const _Float16* vp = Vi + jpos * DH_;
#pragma unroll
      for (int j = 0; j < 4; ++j) oacc[j][e] += p * (float)vp[j * 16 + ln];
    }
  }
#pragma unroll
  for (int j = 0; j < 4; ++j)
#pragma unroll
    for (int e = 0; e < 8; ++e) {
      int i = i0 + r0 + e + hi * 8;
      innerh[(b * NP + TEXT + i) * DIM_ + h * DH_ + j * 16 + ln] = (_Float16)oacc[j][e];
    }
}

// ---------------------------------------------------------------- output GEMM + bias
__global__ void out_gemm(const _Float16* __restrict__ innerh, const _Float16* __restrict__ woT,
                         const float* __restrict__ bout, float* __restrict__ out) {
  int wave = threadIdx.x >> 5, lane = threadIdx.x & 31;
  int hi = lane >> 4, ln = lane & 15;
  int m0 = blockIdx.x * 16;
  int n0 = blockIdx.y * 128 + wave * 16;
  v8f z = {}; v8f acc = z;
  const _Float16* arow = innerh + (m0 + ln) * DIM_ + hi * 8;
  const _Float16* brow = woT + (n0 + ln) * DIM_ + hi * 8;
  for (int kc = 0; kc < DIM_; kc += 32) {
    __builtin_prefetch(arow + kc + 128, 0, 3);
    __builtin_prefetch(brow + kc + 128, 0, 3);
    acc = wmma16(load_a(arow + kc), load_b(brow + kc), acc);
  }
  float bias = bout[n0 + ln];
  int b   = m0 / NP;                   // wave-uniform (1152 % 16 == 0)
  int tt0 = m0 - b * NP;
#pragma unroll
  for (int e = 0; e < 8; ++e) {
    int tt = tt0 + e + hi * 8;
    if (tt < NREAL)
      out[(b * NREAL + tt) * DIM_ + n0 + ln] = acc[e] + bias;
  }
}

// ---------------------------------------------------------------- launch
extern "C" void kernel_launch(void* const* d_in, const int* in_sizes, int n_in,
                              void* d_out, int out_size, void* d_ws, size_t ws_size,
                              hipStream_t stream) {
  const float*         x    = (const float*)d_in[0];
  const unsigned char* mask = (const unsigned char*)d_in[1];
  const float*         Wq   = (const float*)d_in[2];
  const float*         Wo   = (const float*)d_in[3];
  const float*         bo   = (const float*)d_in[4];
  float* out = (float*)d_out;

  _Float16* ws = (_Float16*)d_ws;
  const size_t XN = (size_t)B_ * NP * DIM_;          // 2,359,296
  const size_t QN = (size_t)B_ * H_ * NP * DH_;      // 2,359,296
  _Float16* xh     = ws;
  _Float16* wqTh   = xh + XN;                        // 1536 x 512
  _Float16* woTh   = wqTh + (size_t)1536 * DIM_;     // 512 x 512
  _Float16* qh     = woTh + (size_t)DIM_ * DIM_;
  _Float16* kh     = qh + QN;
  _Float16* vh     = kh + QN;
  _Float16* vTh    = vh + QN;                        // [bh][d][t]
  _Float16* innerh = vTh + QN;

  cvt_x_pad<<<(B_ * NP * DIM_ + 255) / 256, 256, 0, stream>>>(x, xh);
  cvt_wT<<<(DIM_ * 1536 + 255) / 256, 256, 0, stream>>>(Wq, wqTh, DIM_, 1536);
  cvt_wT<<<(DIM_ * DIM_ + 255) / 256, 256, 0, stream>>>(Wo, woTh, DIM_, DIM_);

  qkv_gemm<<<dim3((B_ * NP) / 16, 1536 / 128), 256, 0, stream>>>(xh, wqTh, qh, kh, vh, vTh);
  text_attn<<<B_ * H_, 256, 0, stream>>>(qh, kh, vTh, innerh);
  img_attn<<<dim3(B_ * H_, IMGSEQ / 64), 128, 0, stream>>>(qh, kh, vh, vTh, mask, innerh);
  out_gemm<<<dim3((B_ * NP) / 16, DIM_ / 128), 256, 0, stream>>>(innerh, woTh, bo, out);
}
